// DiTBlock_36842229465899
// MI455X (gfx1250) — matrix operand order repaired and checked by reading, not compiled, not run on hardware
//
#include <hip/hip_runtime.h>
#include <cstdint>
#include <cstring>

// ---------------- problem constants (match reference) ----------------
#define HID    1024
#define HEADS  16
#define HD     64
#define BATCH  2
#define SEQ    1024          // 32 x 32 image tokens
#define BS     (BATCH*SEQ)   // 2048 tokens per stream
#define CTXL   77
#define NE     4
#define INTER  4096

typedef __attribute__((ext_vector_type(16))) __bf16 v16bf;
typedef __attribute__((ext_vector_type(8)))  __bf16 v8bf;
typedef __attribute__((ext_vector_type(4)))  __bf16 v4bf;
typedef __attribute__((ext_vector_type(2)))  __bf16 v2bf;
typedef __attribute__((ext_vector_type(8)))  float  v8f;
typedef __attribute__((ext_vector_type(4)))  int    v4i;

// ---------------- f32 -> bf16 conversion ----------------
#if defined(__AMDGCN__) && __has_builtin(__builtin_amdgcn_cvt_pk_bf16_f32)
__device__ __forceinline__ v2bf cvt2_bf16(float a, float b) {
  auto t = __builtin_amdgcn_cvt_pk_bf16_f32(a, b);   // v_cvt_pk_bf16_f32
  static_assert(sizeof(t) == sizeof(v2bf), "cvt_pk_bf16 size mismatch");
  v2bf r; __builtin_memcpy(&r, &t, sizeof(r));
  return r;
}
__device__ __forceinline__ __bf16 to_bf16(float f) { return cvt2_bf16(f, 0.0f)[0]; }
#else
// native fptrunc: backend legalizes to the HW bf16 cvt on gfx1250
__device__ __forceinline__ __bf16 to_bf16(float f) { return (__bf16)f; }
__device__ __forceinline__ v2bf cvt2_bf16(float a, float b) {
  v2bf r; r[0] = (__bf16)a; r[1] = (__bf16)b; return r;
}
#endif

__device__ __forceinline__ v4bf pack4_bf16(float x, float y, float z, float w) {
  v2bf a = cvt2_bf16(x, y), b = cvt2_bf16(z, w);
  return __builtin_shufflevector(a, b, 0, 1, 2, 3);
}

// ---------------- async global -> LDS copy (CDNA5 path, guarded) ----------------
#if defined(__AMDGCN__) && __has_builtin(__builtin_amdgcn_global_load_async_to_lds_b128)
#define HAS_ASYNC_LDS 1
#else
#define HAS_ASYNC_LDS 0
#endif

__device__ __forceinline__ void async_g2l_b128(const __bf16* g, __bf16* l) {
#if HAS_ASYNC_LDS
  __builtin_amdgcn_global_load_async_to_lds_b128(
      (__attribute__((address_space(1))) v4i*)(v4i*)(void*)g,
      (__attribute__((address_space(3))) v4i*)(v4i*)(void*)l, 0, 0);
#else
  *(v8bf*)l = *(const v8bf*)g;
#endif
}
__device__ __forceinline__ void wait_async_lds() {
#if HAS_ASYNC_LDS
#if __has_builtin(__builtin_amdgcn_s_wait_asynccnt)
  __builtin_amdgcn_s_wait_asynccnt(0);
#else
  asm volatile("s_wait_asynccnt 0" ::: "memory");
#endif
#endif
}

__device__ __forceinline__ v8f zero_v8f() {
  v8f z;
#pragma unroll
  for (int i = 0; i < 8; ++i) z[i] = 0.0f;
  return z;
}

// A-operand fragment (16 x 32 bf16). tile -> (row 0, k 0) of the subtile.
// Lane L holds row m=L&15; elems 0..7 = k 8*hi..+8, elems 8..15 = k 16+8*hi..+8.
__device__ __forceinline__ v16bf frag_A(const __bf16* tile, int stride, int lane) {
  const __bf16* p = tile + (size_t)(lane & 15) * stride + ((lane >> 4) << 3);
  v8bf lo = *(const v8bf*)p;
  v8bf hi = *(const v8bf*)(p + 16);
  return __builtin_shufflevector(lo, hi, 0,1,2,3,4,5,6,7,8,9,10,11,12,13,14,15);
}
// B-operand fragment (32 x 16 bf16): lane L holds column n=L&15;
// elems 0..7 = k 8*hi..+8, elems 8..15 = k 16+8*hi..+8 (matches frag_A).
__device__ __forceinline__ v16bf frag_B(const __bf16* tile, int stride, int lane) {
  const __bf16* p = tile + (size_t)(lane & 15) * stride + ((lane >> 4) << 3);
  v8bf lo = *(const v8bf*)p;
  v8bf hi = *(const v8bf*)(p + 16);
  return __builtin_shufflevector(lo, hi, 0,1,2,3,4,5,6,7,8,9,10,11,12,13,14,15);
}

__device__ __forceinline__ v8f wmma_bf16(v16bf a, v16bf b, v8f c) {
  return __builtin_amdgcn_wmma_f32_16x16x32_bf16(false, a, false, b, (short)0, c, false, false);
}

// ---------------- GEMM: C[N,O] (+)= A[N,K] * W[O,K]^T ----------------
// accumulate==0: C = acc
// accumulate!=0: C = C + acc * (rowscale ? rowscale[row*rs_stride] : 1)
#define KSTEP 64
#define LSTR  72   // LDS row stride in bf16 (144B, 16B aligned, bank-skewed)

__global__ __launch_bounds__(256) void gemm_bf16_wmma(
    const float* __restrict__ A, int lda,
    const float* __restrict__ W,             // [O,K] row-major
    float* __restrict__ C, int ldc,
    const float* __restrict__ rowscale, int rs_stride,
    int N, int K, int O, int accumulate)
{
  __shared__ __bf16 sA[128 * LSTR];
  __shared__ __bf16 sW[128 * LSTR];
  const int tid  = threadIdx.x;
  const int lane = tid & 31;
  const int wid  = tid >> 5;
  const int hi   = lane >> 4;
  const int lm   = lane & 15;
  const int wm   = wid & 3;    // 4 wave-rows x 32
  const int wn   = wid >> 2;   // 2 wave-cols x 64
  const int rowT = blockIdx.y * 128;
  const int colT = blockIdx.x * 128;
  (void)O;

  v8f acc[2][4];
#pragma unroll
  for (int i = 0; i < 2; ++i)
#pragma unroll
    for (int j = 0; j < 4; ++j) acc[i][j] = zero_v8f();

  for (int k0 = 0; k0 < K; k0 += KSTEP) {
    __syncthreads();
    // prefetch next K tile while this one is staged/computed
    if (k0 + KSTEP < K) {
      int r  = tid >> 1;
      int cc = (tid & 1) << 5;   // two 128B halves of 64 floats
      __builtin_prefetch(W + (size_t)(colT + r) * K + k0 + KSTEP + cc, 0, 1);
      int ra = rowT + r;
      if (ra < N) __builtin_prefetch(A + (size_t)ra * lda + k0 + KSTEP + cc, 0, 1);
    }
#pragma unroll
    for (int it = 0; it < 8; ++it) {           // 128 rows x 16 float4 chunks, x2 tiles
      int i  = tid + it * 256;
      int r  = i >> 4;
      int cg = (i & 15) << 2;
      int ra = rowT + r;
      float4 va = make_float4(0.f, 0.f, 0.f, 0.f);
      if (ra < N) va = *(const float4*)(A + (size_t)ra * lda + k0 + cg);
      *(v4bf*)(sA + r * LSTR + cg) = pack4_bf16(va.x, va.y, va.z, va.w);
      float4 vw = *(const float4*)(W + (size_t)(colT + r) * K + k0 + cg);
      *(v4bf*)(sW + r * LSTR + cg) = pack4_bf16(vw.x, vw.y, vw.z, vw.w);
    }
    __syncthreads();

#pragma unroll
    for (int ks = 0; ks < 2; ++ks) {
      v16bf af0 = frag_A(sA + (wm * 32 +  0) * LSTR + ks * 32, LSTR, lane);
      v16bf af1 = frag_A(sA + (wm * 32 + 16) * LSTR + ks * 32, LSTR, lane);
#pragma unroll
      for (int tn = 0; tn < 4; ++tn) {
        v16bf bf = frag_B(sW + (wn * 64 + tn * 16) * LSTR + ks * 32, LSTR, lane);
        acc[0][tn] = wmma_bf16(af0, bf, acc[0][tn]);
        acc[1][tn] = wmma_bf16(af1, bf, acc[1][tn]);
      }
    }
  }

#pragma unroll
  for (int tm = 0; tm < 2; ++tm)
#pragma unroll
    for (int tn = 0; tn < 4; ++tn) {
      int col = colT + wn * 64 + tn * 16 + lm;
#pragma unroll
      for (int v = 0; v < 8; ++v) {
        int row = rowT + wm * 32 + tm * 16 + v + 8 * hi;
        if (row < N) {
          float val = acc[tm][tn][v];
          if (accumulate) {
            float s = rowscale ? rowscale[(size_t)row * rs_stride] : 1.0f;
            val = C[(size_t)row * ldc + col] + val * s;
          }
          C[(size_t)row * ldc + col] = val;
        }
      }
    }
}

// ---------------- LayerNorm (f32 -> f32), one block per token ----------------
__global__ __launch_bounds__(256) void layernorm_kernel(
    const float* __restrict__ x, const float* __restrict__ g,
    const float* __restrict__ b, float* __restrict__ y)
{
  __shared__ float red[256];
  const int n = blockIdx.x, t = threadIdx.x;
  const float* xr = x + (size_t)n * HID;
  float4 v = *(const float4*)(xr + t * 4);
  red[t] = v.x + v.y + v.z + v.w;
  __syncthreads();
  for (int o = 128; o > 0; o >>= 1) { if (t < o) red[t] += red[t + o]; __syncthreads(); }
  float mu = red[0] * (1.0f / HID);
  __syncthreads();
  float dx = v.x - mu, dy = v.y - mu, dz = v.z - mu, dw = v.w - mu;
  red[t] = dx*dx + dy*dy + dz*dz + dw*dw;
  __syncthreads();
  for (int o = 128; o > 0; o >>= 1) { if (t < o) red[t] += red[t + o]; __syncthreads(); }
  float inv = rsqrtf(red[0] * (1.0f / HID) + 1e-5f);
  float4 gg = *(const float4*)(g + t * 4);
  float4 bb = *(const float4*)(b + t * 4);
  float4 out;
  out.x = dx * inv * gg.x + bb.x;
  out.y = dy * inv * gg.y + bb.y;
  out.z = dz * inv * gg.z + bb.z;
  out.w = dw * inv * gg.w + bb.w;
  *(float4*)(y + (size_t)n * HID + t * 4) = out;
}

// ------- head split (+optional 2D RoPE), f32 -> bf16 -------
// transposed==0: dst[B,H,dst_seq,64]   (Q / K layout)
// transposed==1: dst[B,H,64,dst_seq]   (V^T layout for the PV WMMA B-operand)
__global__ __launch_bounds__(256) void split_heads_kernel(
    const float* __restrict__ src, int src_ld, int src_off,
    __bf16* __restrict__ dst, int S_src, int S_chunk,
    int dst_off, int dst_seq, int do_rope, int transposed)
{
  int idx = blockIdx.x * blockDim.x + threadIdx.x;
  int c = idx & 31;
  int s = (idx >> 5) % S_chunk;
  int hh = ((idx >> 5) / S_chunk) % HEADS;
  int b = (idx >> 5) / (S_chunk * HEADS);
  if (b >= BATCH) return;
  float o1 = 0.0f, o2 = 0.0f;
  if (s < S_src) {
    const float* p = src + (size_t)(b * S_src + s) * src_ld + src_off + hh * HD;
    float x1 = p[c], x2 = p[c + 32];
    if (do_rope) {
      int row = s >> 5, col = s & 31;                       // width = 32
      float f  = __powf(10000.0f, -(float)(c & 15) * (1.0f / 16.0f));
      float a1 = (float)col * f;                            // x-channel (c < 32)
      float a2 = (float)row * f;                            // y-channel (c + 32)
      o1 = x1 * __cosf(a1) - x2 * __sinf(a1);
      o2 = x2 * __cosf(a2) + x1 * __sinf(a2);
    } else { o1 = x1; o2 = x2; }
  }
  size_t base = (size_t)(b * HEADS + hh) * HD * dst_seq;
  if (transposed) {
    dst[base + (size_t)c        * dst_seq + dst_off + s] = to_bf16(o1);
    dst[base + (size_t)(c + 32) * dst_seq + dst_off + s] = to_bf16(o2);
  } else {
    __bf16* d = dst + base + (size_t)(dst_off + s) * HD;
    d[c]      = to_bf16(o1);
    d[c + 32] = to_bf16(o2);
  }
}

// ---------------- flash attention, WMMA bf16, online softmax f32 ----------------
// Q [B,H,Sq,64] bf16, K [B,H,Skv,64] bf16, Vt [B,H,64,Skv] bf16 (transposed,
// zero-padded past kv_len; Skv multiple of 128). Out [B*Sq, out_ld] f32 merged.
#define KLSTR 72
__global__ __launch_bounds__(128) void flash_attn_wmma(
    const __bf16* __restrict__ Q, const __bf16* __restrict__ Kc,
    const __bf16* __restrict__ Vt, float* __restrict__ Out, int out_ld,
    int Sq, int Skv, int kv_len, float scale)
{
  __shared__ __bf16 sK[128 * KLSTR];      // K tile: [key][d], async-staged
  __shared__ __bf16 sP[4 * 16 * 136];     // per-wave P tile 16 x 128
  const int tid = threadIdx.x, lane = tid & 31, w = tid >> 5;
  const int hi = lane >> 4, lm = lane & 15;
  const int b = blockIdx.z, h = blockIdx.y;
  const int q0 = blockIdx.x * 64 + w * 16;
  const __bf16* qb = Q  + (size_t)(b * HEADS + h) * Sq  * HD;
  const __bf16* kb = Kc + (size_t)(b * HEADS + h) * Skv * HD;
  const __bf16* vb = Vt + (size_t)(b * HEADS + h) * HD * Skv;   // [64][Skv]
  __bf16* pw = sP + w * 16 * 136;

  // Q fragments are reused across all key tiles
  v16bf qf0 = frag_A(qb + (size_t)q0 * HD +  0, HD, lane);
  v16bf qf1 = frag_A(qb + (size_t)q0 * HD + 32, HD, lane);

  v8f oacc[4];
#pragma unroll
  for (int j = 0; j < 4; ++j) oacc[j] = zero_v8f();
  float rm[8], rl[8];
#pragma unroll
  for (int v = 0; v < 8; ++v) { rm[v] = -3.0e38f; rl[v] = 0.0f; }

  const int nkt = Skv >> 7;
  for (int kt = 0; kt < nkt; ++kt) {
    __syncthreads();
    // stage K tile (128 keys x 64 d = 16KB) via async global->LDS b128 copies
#pragma unroll
    for (int it = 0; it < 8; ++it) {
      int cidx = tid + it * 128;           // 1024 chunks of 16B
      int r  = cidx >> 3;                  // key row 0..127
      int co = (cidx & 7) << 3;            // d offset 0..56 step 8
      async_g2l_b128(kb + (size_t)(kt * 128 + r) * HD + co, sK + r * KLSTR + co);
    }
    wait_async_lds();
    __syncthreads();

    // scores: 16 x 128 from LDS K tile
    v8f sc[8];
#pragma unroll
    for (int t = 0; t < 8; ++t) sc[t] = zero_v8f();
#pragma unroll
    for (int t = 0; t < 8; ++t) {
      const __bf16* kt_l = sK + (t * 16) * KLSTR;
      sc[t] = wmma_bf16(qf0, frag_B(kt_l +  0, KLSTR, lane), sc[t]);
      sc[t] = wmma_bf16(qf1, frag_B(kt_l + 32, KLSTR, lane), sc[t]);
    }

    float tmax[8];
#pragma unroll
    for (int v = 0; v < 8; ++v) tmax[v] = -3.0e38f;
#pragma unroll
    for (int t = 0; t < 8; ++t) {
      int key = kt * 128 + t * 16 + lm;
      bool ok = key < kv_len;
#pragma unroll
      for (int v = 0; v < 8; ++v) {
        float sv = ok ? sc[t][v] * scale : -3.0e38f;
        sc[t][v] = sv;
        tmax[v] = fmaxf(tmax[v], sv);
      }
    }
#pragma unroll
    for (int v = 0; v < 8; ++v)
#pragma unroll
      for (int off = 1; off < 16; off <<= 1)
        tmax[v] = fmaxf(tmax[v], __shfl_xor(tmax[v], off, 32));

    float alpha[8];
#pragma unroll
    for (int v = 0; v < 8; ++v) {
      float mn = fmaxf(rm[v], tmax[v]);
      alpha[v] = __expf(rm[v] - mn);
      rm[v] = mn;
    }
#pragma unroll
    for (int tn = 0; tn < 4; ++tn)
#pragma unroll
      for (int v = 0; v < 8; ++v) oacc[tn][v] *= alpha[v];

    float rs[8] = {0.f,0.f,0.f,0.f,0.f,0.f,0.f,0.f};
#pragma unroll
    for (int t = 0; t < 8; ++t)
#pragma unroll
      for (int v = 0; v < 8; ++v) {
        float p = __expf(sc[t][v] - rm[v]);
        rs[v] += p;
        pw[(v + 8 * hi) * 136 + t * 16 + lm] = to_bf16(p);   // re-layout via LDS
      }
#pragma unroll
    for (int v = 0; v < 8; ++v) {
#pragma unroll
      for (int off = 1; off < 16; off <<= 1) rs[v] += __shfl_xor(rs[v], off, 32);
      rl[v] = rl[v] * alpha[v] + rs[v];
    }

    // P (16x128) x V (128x64); V^T rows are key-contiguous -> direct global B-frags
#pragma unroll
    for (int kk = 0; kk < 4; ++kk) {
      v16bf pf = frag_A(pw + kk * 32, 136, lane);
#pragma unroll
      for (int tn = 0; tn < 4; ++tn)
        oacc[tn] = wmma_bf16(
            pf, frag_B(vb + (size_t)(tn * 16) * Skv + kt * 128 + kk * 32, Skv, lane),
            oacc[tn]);
    }
  }

#pragma unroll
  for (int tn = 0; tn < 4; ++tn)
#pragma unroll
    for (int v = 0; v < 8; ++v) {
      int row = b * Sq + q0 + v + 8 * hi;
      Out[(size_t)row * out_ld + h * HD + tn * 16 + lm] = oacc[tn][v] / rl[v];
    }
}

// ---------------- MoE gate: softmax over 4 logits, top-2, renormalize ----------------
__global__ __launch_bounds__(128) void gate_topk_kernel(
    const float* __restrict__ xn, const float* __restrict__ gate, float* __restrict__ wout)
{
  __shared__ float red[NE][128];
  const int n = blockIdx.x, t = threadIdx.x;
  const float* xr = xn + (size_t)n * HID;
  float a[NE] = {0.f, 0.f, 0.f, 0.f};
  for (int hh = t; hh < HID; hh += 128) {
    float xv = xr[hh];
#pragma unroll
    for (int e = 0; e < NE; ++e) a[e] += xv * gate[e * HID + hh];
  }
#pragma unroll
  for (int e = 0; e < NE; ++e) red[e][t] = a[e];
  __syncthreads();
  for (int o = 64; o > 0; o >>= 1) {
    if (t < o) {
#pragma unroll
      for (int e = 0; e < NE; ++e) red[e][t] += red[e][t + o];
    }
    __syncthreads();
  }
  if (t == 0) {
    float p[NE], mx = -3.0e38f, sum = 0.f;
    for (int e = 0; e < NE; ++e) mx = fmaxf(mx, red[e][0]);
    for (int e = 0; e < NE; ++e) { p[e] = __expf(red[e][0] - mx); sum += p[e]; }
    for (int e = 0; e < NE; ++e) p[e] /= sum;
    int i1 = 0;
    for (int e = 1; e < NE; ++e) if (p[e] > p[i1]) i1 = e;
    int i2 = (i1 == 0) ? 1 : 0;
    for (int e = 0; e < NE; ++e) if (e != i1 && p[e] > p[i2]) i2 = e;
    float s2 = p[i1] + p[i2] + 1e-5f;
    for (int e = 0; e < NE; ++e)
      wout[(size_t)n * NE + e] = (e == i1 || e == i2) ? p[e] / s2 : 0.0f;
  }
}

// ---------------- h = silu(g) * u, in-place into g ----------------
__global__ __launch_bounds__(256) void silu_mul_kernel(
    float* __restrict__ g, const float* __restrict__ u, int n)
{
  int i = blockIdx.x * blockDim.x + threadIdx.x;
  if (i < n) {
    float gv = g[i];
    g[i] = gv / (1.0f + __expf(-gv)) * u[i];
  }
}

// ---------------- host-side orchestration ----------------
static inline void launch_gemm(hipStream_t st, const float* A, int lda, const float* W,
                               float* C, int ldc, const float* rs, int rss,
                               int N, int K, int O, int acc) {
  dim3 g(O / 128, (N + 127) / 128);
  gemm_bf16_wmma<<<g, 256, 0, st>>>(A, lda, W, C, ldc, rs, rss, N, K, O, acc);
}
static inline void launch_split(hipStream_t st, const float* src, int sld, int soff,
                                __bf16* dst, int Ssrc, int Schunk, int doff, int dseq,
                                int rope, int transposed) {
  int threads = BATCH * HEADS * Schunk * 32;
  split_heads_kernel<<<threads / 256, 256, 0, st>>>(src, sld, soff, dst, Ssrc, Schunk,
                                                    doff, dseq, rope, transposed);
}

extern "C" void kernel_launch(void* const* d_in, const int* in_sizes, int n_in,
                              void* d_out, int out_size, void* d_ws, size_t ws_size,
                              hipStream_t stream)
{
  (void)in_sizes; (void)n_in; (void)out_size; (void)ws_size;
  auto F = [&](int i) { return (const float*)d_in[i]; };
  // Input order: x_a(0), x_b(1), context(2), then params flattened with
  // sorted dict keys (JAX tree order):
  //  cross_a: 3 norm_b, 4 norm_g, 5 wk, 6 wo, 7 wq, 8 wv
  //  cross_b: 9..14 (same order)
  //  dual:   15 norm_a_b, 16 norm_a_g, 17 norm_b_b, 18 norm_b_g,
  //          19 out_a, 20 out_b, 21 qkv_a, 22 qkv_b
  //  moe_a:  23 dw, 24 gate, 25 gw, 26 r_norm_b, 27 r_norm_g,
  //          28 s_dw, 29 s_gw, 30 s_uw, 31 uw
  //  moe_b:  32..40 (same order); 41 height, 42 width (unused; constants)
  const float* ctx = F(2);

  // ---- workspace carve (all 256B aligned) ----
  char* wsp = (char*)d_ws;
  auto take = [&](size_t bytes) { void* p = (void*)wsp; wsp += (bytes + 255) & ~(size_t)255; return p; };
  float*  buf_ln = (float*) take((size_t)BS * HID * 4);
  float*  qkvA   = (float*) take((size_t)BS * 3 * HID * 4);
  float*  qkvB   = (float*) take((size_t)BS * 3 * HID * 4);
  __bf16* q_bf   = (__bf16*)take((size_t)BATCH * HEADS * SEQ * HD * 2);
  __bf16* kc_bf  = (__bf16*)take((size_t)BATCH * HEADS * 2 * SEQ * HD * 2);
  __bf16* vc_bf  = (__bf16*)take((size_t)BATCH * HEADS * 2 * SEQ * HD * 2);  // [B,H,64,2S]
  float*  attn_o = (float*) take((size_t)BS * HID * 4);
  float*  ctxk   = (float*) take((size_t)BATCH * CTXL * HID * 4);
  float*  ctxv   = (float*) take((size_t)BATCH * CTXL * HID * 4);
  __bf16* kx_bf  = (__bf16*)take((size_t)BATCH * HEADS * 128 * HD * 2);
  __bf16* vx_bf  = (__bf16*)take((size_t)BATCH * HEADS * 128 * HD * 2);      // [B,H,64,128]
  float*  gatew  = (float*) take((size_t)BS * NE * 4);
  float*  gbuf   = (float*) take((size_t)BS * INTER * 4);
  float*  ubuf   = (float*) take((size_t)BS * INTER * 4);
  float*  xcopy  = (float*) take((size_t)BS * HID * 4);

  float* xa = (float*)d_out;
  float* xb = xa + (size_t)BS * HID;
  (void)hipMemcpyAsync(xa, d_in[0], (size_t)BS * HID * 4, hipMemcpyDeviceToDevice, stream);
  (void)hipMemcpyAsync(xb, d_in[1], (size_t)BS * HID * 4, hipMemcpyDeviceToDevice, stream);

  const float scale = 0.125f;  // HD^-0.5

  // ================= dual attention (joint K/V over both streams) =================
  layernorm_kernel<<<BS, 256, 0, stream>>>(xa, F(16), F(15), buf_ln);
  launch_gemm(stream, buf_ln, HID, F(21), qkvA, 3 * HID, nullptr, 0, BS, HID, 3 * HID, 0);
  layernorm_kernel<<<BS, 256, 0, stream>>>(xb, F(18), F(17), buf_ln);
  launch_gemm(stream, buf_ln, HID, F(22), qkvB, 3 * HID, nullptr, 0, BS, HID, 3 * HID, 0);

  launch_split(stream, qkvA, 3 * HID, 0,       q_bf,  SEQ, SEQ, 0,   SEQ,     1, 0); // q_a +RoPE
  launch_split(stream, qkvA, 3 * HID, HID,     kc_bf, SEQ, SEQ, 0,   2 * SEQ, 1, 0); // k_a
  launch_split(stream, qkvB, 3 * HID, HID,     kc_bf, SEQ, SEQ, SEQ, 2 * SEQ, 1, 0); // k_b
  launch_split(stream, qkvA, 3 * HID, 2 * HID, vc_bf, SEQ, SEQ, 0,   2 * SEQ, 0, 1); // v_a (T)
  launch_split(stream, qkvB, 3 * HID, 2 * HID, vc_bf, SEQ, SEQ, SEQ, 2 * SEQ, 0, 1); // v_b (T)

  flash_attn_wmma<<<dim3(SEQ / 64, HEADS, BATCH), 128, 0, stream>>>(
      q_bf, kc_bf, vc_bf, attn_o, HID, SEQ, 2 * SEQ, 2 * SEQ, scale);
  launch_gemm(stream, attn_o, HID, F(19), xa, HID, nullptr, 0, BS, HID, HID, 1);  // x_a += out

  launch_split(stream, qkvB, 3 * HID, 0, q_bf, SEQ, SEQ, 0, SEQ, 1, 0);           // q_b +RoPE
  flash_attn_wmma<<<dim3(SEQ / 64, HEADS, BATCH), 128, 0, stream>>>(
      q_bf, kc_bf, vc_bf, attn_o, HID, SEQ, 2 * SEQ, 2 * SEQ, scale);
  launch_gemm(stream, attn_o, HID, F(20), xb, HID, nullptr, 0, BS, HID, HID, 1);  // x_b += out

  // ================= cross attention =================
  for (int sidx = 0; sidx < 2; ++sidx) {
    float* x = sidx ? xb : xa;
    int base = sidx ? 9 : 3;
    const float* nb = F(base + 0); const float* ng = F(base + 1);
    const float* wk = F(base + 2); const float* wo = F(base + 3);
    const float* wq = F(base + 4); const float* wv = F(base + 5);

    layernorm_kernel<<<BS, 256, 0, stream>>>(x, ng, nb, buf_ln);
    launch_gemm(stream, buf_ln, HID, wq, attn_o, HID, nullptr, 0, BS, HID, HID, 0);
    launch_gemm(stream, ctx, HID, wk, ctxk, HID, nullptr, 0, BATCH * CTXL, HID, HID, 0);
    launch_gemm(stream, ctx, HID, wv, ctxv, HID, nullptr, 0, BATCH * CTXL, HID, HID, 0);
    launch_split(stream, attn_o, HID, 0, q_bf,  SEQ,  SEQ, 0, SEQ, 0, 0);
    launch_split(stream, ctxk,   HID, 0, kx_bf, CTXL, 128, 0, 128, 0, 0);  // zero-pad
    launch_split(stream, ctxv,   HID, 0, vx_bf, CTXL, 128, 0, 128, 0, 1);  // V^T, zero-pad
    flash_attn_wmma<<<dim3(SEQ / 64, HEADS, BATCH), 128, 0, stream>>>(
        q_bf, kx_bf, vx_bf, attn_o, HID, SEQ, 128, CTXL, scale);
    launch_gemm(stream, attn_o, HID, wo, x, HID, nullptr, 0, BS, HID, HID, 1);
  }

  // ================= MoE (dense over experts, like the reference einsum) =================
  for (int sidx = 0; sidx < 2; ++sidx) {
    float* x = sidx ? xb : xa;
    int base = sidx ? 32 : 23;
    const float* dwp  = F(base + 0); const float* gate = F(base + 1);
    const float* gw   = F(base + 2); const float* rnb  = F(base + 3);
    const float* rng  = F(base + 4); const float* sdw  = F(base + 5);
    const float* sgw  = F(base + 6); const float* suw  = F(base + 7);
    const float* uw   = F(base + 8);

    (void)hipMemcpyAsync(xcopy, x, (size_t)BS * HID * 4, hipMemcpyDeviceToDevice, stream);
    layernorm_kernel<<<BS, 256, 0, stream>>>(xcopy, rng, rnb, buf_ln);
    gate_topk_kernel<<<BS, 128, 0, stream>>>(buf_ln, gate, gatew);

    // shared expert
    launch_gemm(stream, xcopy, HID, sgw, gbuf, INTER, nullptr, 0, BS, HID, INTER, 0);
    launch_gemm(stream, xcopy, HID, suw, ubuf, INTER, nullptr, 0, BS, HID, INTER, 0);
    silu_mul_kernel<<<(BS * INTER) / 256, 256, 0, stream>>>(gbuf, ubuf, BS * INTER);
    launch_gemm(stream, gbuf, INTER, sdw, x, HID, nullptr, 0, BS, INTER, HID, 1);

    // routed experts, weighted by top-2 gate coefficients
    for (int e = 0; e < NE; ++e) {
      launch_gemm(stream, xcopy, HID, gw + (size_t)e * INTER * HID, gbuf, INTER,
                  nullptr, 0, BS, HID, INTER, 0);
      launch_gemm(stream, xcopy, HID, uw + (size_t)e * INTER * HID, ubuf, INTER,
                  nullptr, 0, BS, HID, INTER, 0);
      silu_mul_kernel<<<(BS * INTER) / 256, 256, 0, stream>>>(gbuf, ubuf, BS * INTER);
      launch_gemm(stream, gbuf, INTER, dwp + (size_t)e * HID * INTER, x, HID,
                  gatew + e, NE, BS, INTER, HID, 1);
    }
  }
}
